// MPNNLayer_23235773072079
// MI455X (gfx1250) — matrix operand order repaired and checked by reading, not compiled, not run on hardware
//
#include <hip/hip_runtime.h>

#define H 128          // hidden size
#define HT (H / 16)    // 8 n-tiles

typedef __attribute__((ext_vector_type(16))) _Float16 v16h;
typedef __attribute__((ext_vector_type(8)))  float    v8f;

union V16H {
    v16h v;
    _Float16 h[16];
    uint4 q[2];
};

struct RawA { float4 a, b, c, d; };

// Fast SiLU: v * v_rcp_f32(1 + exp(-v)).
__device__ __forceinline__ float silu_f(float v) {
    float e = __expf(-v);
    return v * __builtin_amdgcn_rcpf(1.0f + e);
}

// Issue the 4 global b128 loads for one 16x32 A tile row-slice.
__device__ __forceinline__ RawA load_raw(const float* __restrict__ rowp, int hi) {
    const float4* rp = (const float4*)rowp;
    RawA r;
    r.a = rp[hi * 2];
    r.b = rp[hi * 2 + 1];
    r.c = rp[4 + hi * 2];
    r.d = rp[4 + hi * 2 + 1];
    return r;
}

// Convert raw f32 quads to the f16 WMMA A-operand register layout.
__device__ __forceinline__ v16h cvt_raw(const RawA& r) {
    V16H A;
    A.h[0]  = (_Float16)r.a.x; A.h[1]  = (_Float16)r.a.y;
    A.h[2]  = (_Float16)r.a.z; A.h[3]  = (_Float16)r.a.w;
    A.h[4]  = (_Float16)r.b.x; A.h[5]  = (_Float16)r.b.y;
    A.h[6]  = (_Float16)r.b.z; A.h[7]  = (_Float16)r.b.w;
    A.h[8]  = (_Float16)r.c.x; A.h[9]  = (_Float16)r.c.y;
    A.h[10] = (_Float16)r.c.z; A.h[11] = (_Float16)r.c.w;
    A.h[12] = (_Float16)r.d.x; A.h[13] = (_Float16)r.d.y;
    A.h[14] = (_Float16)r.d.z; A.h[15] = (_Float16)r.d.w;
    return A.v;
}

__device__ __forceinline__ void load_b(V16H& B, const char* lw, int kt, int nt, int lane) {
    const char* bp = lw + (kt * 8 + nt) * 1024 + lane * 32;
    B.q[0] = *(const uint4*)(bp);
    B.q[1] = *(const uint4*)(bp + 16);
}

// ---------------------------------------------------------------------------
__global__ void zero_f32(float* __restrict__ p, int n) {
    int i = blockIdx.x * blockDim.x + threadIdx.x;
    if (i < n) p[i] = 0.0f;
}

// ---------------------------------------------------------------------------
// f32 W (Ktot x 128, row-major) -> f16 pre-swizzled WMMA B-layout tiles.
// Tile (kt,nt) = 1024 B; lane L's 16 halves contiguous at L*32.
// B layout: lane L holds N = L&15, K = (j>>3)*16 + (L>=16?8:0) + (j&7).
// ---------------------------------------------------------------------------
__global__ void swizzle_w(const float* __restrict__ W, _Float16* __restrict__ out, int Ktot) {
    int t = blockIdx.x * blockDim.x + threadIdx.x;
    int total = Ktot * H;
    if (t >= total) return;
    int tile = t >> 9;
    int r    = t & 511;
    int L    = r >> 4;
    int j    = r & 15;
    int kt   = tile >> 3;
    int nt   = tile & 7;
    int hi   = L >> 4;
    int n    = L & 15;
    int k    = kt * 32 + (j >> 3) * 16 + hi * 8 + (j & 7);
    out[t] = (_Float16)W[(size_t)k * H + nt * 16 + n];
}

// ---------------------------------------------------------------------------
// Edge kernel, M=32 per wave (two 16-edge sub-tiles sharing every B tile),
// software-pipelined A (global) and B (LDS) operand streams.
// LDS: w1 64KB | w2 32KB | h staging 8 waves * 8KB = 64KB  (= 160KB)
// ---------------------------------------------------------------------------
__global__ void __launch_bounds__(256)
edge_kernel(const float* __restrict__ x, const int* __restrict__ ei,
            const float* __restrict__ ew, const float* __restrict__ mW1,
            const float* __restrict__ mb1, const float* __restrict__ mb2,
            const _Float16* __restrict__ w1s, const _Float16* __restrict__ w2s,
            float* __restrict__ agg, int E) {
    extern __shared__ char smem[];
    char* lw1 = smem;               // 65536 B
    char* lw2 = smem + 65536;       // 32768 B
    char* lh  = smem + 98304;       // 65536 B (8 waves * 8KB)

    {
        const uint4* g1 = (const uint4*)w1s;
        const uint4* g2 = (const uint4*)w2s;
        uint4* s1 = (uint4*)lw1;
        uint4* s2 = (uint4*)lw2;
        for (int i = threadIdx.x; i < 4096; i += 256) s1[i] = g1[i];
        for (int i = threadIdx.x; i < 2048; i += 256) s2[i] = g2[i];
    }
    __syncthreads();

    const int lane = threadIdx.x & 31;
    const int wave = threadIdx.x >> 5;
    const int hi   = lane >> 4;
    const int n0   = lane & 15;

    const int numTiles = E >> 5;    // 32 edges per tile
    for (int tile = blockIdx.x * 8 + wave; tile < numTiles; tile += gridDim.x * 8) {
        const int e0 = tile * 32;

        int   dstA[2][8];
        float ewA[2][8];
        const float* rows[2][2];    // [sub][src/dst]
#pragma unroll
        for (int s = 0; s < 2; ++s) {
            rows[s][0] = x + (size_t)ei[e0 + s * 16 + n0] * H;
            rows[s][1] = x + (size_t)ei[E + e0 + s * 16 + n0] * H;
#pragma unroll
            for (int i = 0; i < 8; ++i) {
                int m = s * 16 + hi * 8 + i;
                dstA[s][i] = ei[E + e0 + m];
                ewA[s][i]  = ew[e0 + m];
            }
        }

        // ---- GEMM1: bias + ew column folded into C init ----
        v8f acc[2][HT];
#pragma unroll
        for (int nt = 0; nt < HT; ++nt) {
            int n = nt * 16 + n0;
            float bias = mb1[n];
            float wl   = mW1[256 * H + n];
#pragma unroll
            for (int s = 0; s < 2; ++s)
#pragma unroll
                for (int i = 0; i < 8; ++i)
                    acc[s][nt][i] = bias + ewA[s][i] * wl;
        }

        // pipelined over kt: raw A for kt+1 prefetched, B double-buffered
        RawA r0 = load_raw(rows[0][0], hi);
        RawA r1 = load_raw(rows[1][0], hi);
#pragma unroll
        for (int kt = 0; kt < 8; ++kt) {
            v16h A0 = cvt_raw(r0);
            v16h A1 = cvt_raw(r1);
            if (kt < 7) {
                int kn = kt + 1;
                const float* p0 = rows[0][kn >> 2] + (kn & 3) * 32;
                const float* p1 = rows[1][kn >> 2] + (kn & 3) * 32;
                r0 = load_raw(p0, hi);
                r1 = load_raw(p1, hi);
            }
            V16H Bb[2];
            load_b(Bb[0], lw1, kt, 0, lane);
#pragma unroll
            for (int nt = 0; nt < HT; ++nt) {
                if (nt < HT - 1) load_b(Bb[(nt + 1) & 1], lw1, kt, nt + 1, lane);
                v16h Bv = Bb[nt & 1].v;
                acc[0][nt] = __builtin_amdgcn_wmma_f32_16x16x32_f16(
                    false, A0, false, Bv, (short)0, acc[0][nt], false, false);
                acc[1][nt] = __builtin_amdgcn_wmma_f32_16x16x32_f16(
                    false, A1, false, Bv, (short)0, acc[1][nt], false, false);
            }
        }

        // ---- SiLU -> f16 -> LDS staging (both sub-tiles) ----
        char* hbase = lh + wave * 8192;
#pragma unroll
        for (int s = 0; s < 2; ++s)
#pragma unroll
            for (int nt = 0; nt < HT; ++nt)
#pragma unroll
                for (int i = 0; i < 8; ++i) {
                    int m = hi * 8 + i;
                    int n = nt * 16 + n0;
                    *(_Float16*)(hbase + s * 4096 + (m * H + n) * 2) =
                        (_Float16)silu_f(acc[s][nt][i]);
                }
        asm volatile("s_wait_dscnt 0" ::: "memory");

        // ---- GEMM2 (B tiles shared across the two sub-tiles) ----
        v8f acc2[2][HT];
#pragma unroll
        for (int nt = 0; nt < HT; ++nt) {
            float bias = mb2[nt * 16 + n0];
#pragma unroll
            for (int s = 0; s < 2; ++s)
#pragma unroll
                for (int i = 0; i < 8; ++i) acc2[s][nt][i] = bias;
        }
#pragma unroll
        for (int kt = 0; kt < 4; ++kt) {
            V16H A0, A1;
            const char* ab0 = hbase + (n0 * H + kt * 32) * 2;
            const char* ab1 = ab0 + 4096;
            A0.q[0] = *(const uint4*)(ab0 + hi * 16);
            A0.q[1] = *(const uint4*)(ab0 + 32 + hi * 16);
            A1.q[0] = *(const uint4*)(ab1 + hi * 16);
            A1.q[1] = *(const uint4*)(ab1 + 32 + hi * 16);
            V16H Bb[2];
            load_b(Bb[0], lw2, kt, 0, lane);
#pragma unroll
            for (int nt = 0; nt < HT; ++nt) {
                if (nt < HT - 1) load_b(Bb[(nt + 1) & 1], lw2, kt, nt + 1, lane);
                v16h Bv = Bb[nt & 1].v;
                acc2[0][nt] = __builtin_amdgcn_wmma_f32_16x16x32_f16(
                    false, A0.v, false, Bv, (short)0, acc2[0][nt], false, false);
                acc2[1][nt] = __builtin_amdgcn_wmma_f32_16x16x32_f16(
                    false, A1.v, false, Bv, (short)0, acc2[1][nt], false, false);
            }
        }

        // ---- scale by ew, scatter-add into agg (L2-resident) ----
#pragma unroll
        for (int s = 0; s < 2; ++s)
#pragma unroll
            for (int nt = 0; nt < HT; ++nt) {
                int n = nt * 16 + n0;
#pragma unroll
                for (int i = 0; i < 8; ++i) {
                    unsafeAtomicAdd(&agg[(size_t)dstA[s][i] * H + n],
                                    acc2[s][nt][i] * ewA[s][i]);
                }
            }
    }
}

// ---------------------------------------------------------------------------
// Node kernel: u = silu(LN([x, agg] @ uW1 + ub1));  out = x + u @ uW2 + ub2
// LDS: u1 64KB | u2 32KB | f32 staging 64KB | stats 1KB
// ---------------------------------------------------------------------------
__global__ void __launch_bounds__(256)
node_kernel(const float* __restrict__ x, const float* __restrict__ agg,
            const float* __restrict__ ub1, const float* __restrict__ ln_g,
            const float* __restrict__ ln_b, const float* __restrict__ ub2,
            const _Float16* __restrict__ u1s, const _Float16* __restrict__ u2s,
            float* __restrict__ out, int N) {
    extern __shared__ char smem[];
    char*  lu1 = smem;                       // 65536 B
    char*  lu2 = smem + 65536;               // 32768 B
    char*  lf  = smem + 98304;               // 65536 B
    float* lst = (float*)(smem + 98304 + 65536);

    {
        const uint4* g1 = (const uint4*)u1s;
        const uint4* g2 = (const uint4*)u2s;
        uint4* s1 = (uint4*)lu1;
        uint4* s2 = (uint4*)lu2;
        for (int i = threadIdx.x; i < 4096; i += 256) s1[i] = g1[i];
        for (int i = threadIdx.x; i < 2048; i += 256) s2[i] = g2[i];
    }
    __syncthreads();

    const int lane = threadIdx.x & 31;
    const int wave = threadIdx.x >> 5;
    const int hi   = lane >> 4;
    const int n0   = lane & 15;

    const int numTiles = N >> 4;
    for (int tile = blockIdx.x * 8 + wave; tile < numTiles; tile += gridDim.x * 8) {
        const int t0   = tile * 16;
        const int rowA = t0 + n0;
        const float* rows[2] = { x + (size_t)rowA * H, agg + (size_t)rowA * H };

        v8f acc[HT];
#pragma unroll
        for (int nt = 0; nt < HT; ++nt) {
            float bias = ub1[nt * 16 + n0];
#pragma unroll
            for (int i = 0; i < 8; ++i) acc[nt][i] = bias;
        }

        RawA r0 = load_raw(rows[0], hi);
#pragma unroll
        for (int kt = 0; kt < 8; ++kt) {
            v16h A = cvt_raw(r0);
            if (kt < 7) {
                int kn = kt + 1;
                r0 = load_raw(rows[kn >> 2] + (kn & 3) * 32, hi);
            }
            V16H Bb[2];
            load_b(Bb[0], lu1, kt, 0, lane);
#pragma unroll
            for (int nt = 0; nt < HT; ++nt) {
                if (nt < HT - 1) load_b(Bb[(nt + 1) & 1], lu1, kt, nt + 1, lane);
                acc[nt] = __builtin_amdgcn_wmma_f32_16x16x32_f16(
                    false, A, false, Bb[nt & 1].v, (short)0, acc[nt], false, false);
            }
        }

        // ---- stage f32 pre-LN activations to LDS ----
        float* hb = (float*)(lf + wave * 8192);
#pragma unroll
        for (int nt = 0; nt < HT; ++nt)
#pragma unroll
            for (int i = 0; i < 8; ++i)
                hb[(hi * 8 + i) * H + nt * 16 + n0] = acc[nt][i];
        asm volatile("s_wait_dscnt 0" ::: "memory");

        // ---- per-row LayerNorm stats (lanes 0..15, one row each) ----
        float* st = lst + wave * 32;
        if (lane < 16) {
            float s = 0.0f, ss = 0.0f;
            const float4* r4 = (const float4*)(hb + lane * H);
#pragma unroll
            for (int q = 0; q < 32; ++q) {
                float4 f = r4[q];
                s  += f.x + f.y + f.z + f.w;
                ss += f.x * f.x + f.y * f.y + f.z * f.z + f.w * f.w;
            }
            float mu  = s * (1.0f / H);
            float var = ss * (1.0f / H) - mu * mu;
            st[lane]      = mu;
            st[16 + lane] = rsqrtf(var + 1e-5f);
        }
        asm volatile("s_wait_dscnt 0" ::: "memory");

        const float mu   = st[n0];
        const float rstd = st[16 + n0];

        // ---- GEMM2 with LN+SiLU fused into A build ----
        v8f acc2[HT];
#pragma unroll
        for (int nt = 0; nt < HT; ++nt) {
            float bias = ub2[nt * 16 + n0];
#pragma unroll
            for (int i = 0; i < 8; ++i) acc2[nt][i] = bias;
        }
#pragma unroll
        for (int kt = 0; kt < 4; ++kt) {
            V16H A;
#pragma unroll
            for (int j = 0; j < 16; ++j) {
                int k  = kt * 32 + (j >> 3) * 16 + hi * 8 + (j & 7);
                float v = hb[n0 * H + k];
                float t = (v - mu) * rstd * ln_g[k] + ln_b[k];
                A.h[j] = (_Float16)silu_f(t);
            }
            V16H Bb[2];
            load_b(Bb[0], lu2, kt, 0, lane);
#pragma unroll
            for (int nt = 0; nt < HT; ++nt) {
                if (nt < HT - 1) load_b(Bb[(nt + 1) & 1], lu2, kt, nt + 1, lane);
                acc2[nt] = __builtin_amdgcn_wmma_f32_16x16x32_f16(
                    false, A.v, false, Bb[nt & 1].v, (short)0, acc2[nt], false, false);
            }
        }

        // ---- residual + store ----
#pragma unroll
        for (int nt = 0; nt < HT; ++nt) {
            int n = nt * 16 + n0;
#pragma unroll
            for (int i = 0; i < 8; ++i) {
                size_t idx = (size_t)(t0 + hi * 8 + i) * H + n;
                out[idx] = x[idx] + acc2[nt][i];
            }
        }
    }
}

// ---------------------------------------------------------------------------
extern "C" void kernel_launch(void* const* d_in, const int* in_sizes, int n_in,
                              void* d_out, int out_size, void* d_ws, size_t ws_size,
                              hipStream_t stream) {
    const float* x    = (const float*)d_in[0];
    const int*   ei   = (const int*)d_in[1];
    const float* ew   = (const float*)d_in[2];
    const float* mW1  = (const float*)d_in[3];
    const float* mb1  = (const float*)d_in[4];
    const float* mW2  = (const float*)d_in[5];
    const float* mb2  = (const float*)d_in[6];
    const float* uW1  = (const float*)d_in[7];
    const float* ub1  = (const float*)d_in[8];
    const float* ln_g = (const float*)d_in[9];
    const float* ln_b = (const float*)d_in[10];
    const float* uW2  = (const float*)d_in[11];
    const float* ub2  = (const float*)d_in[12];
    float* out = (float*)d_out;

    const int N = in_sizes[0] / H;     // 10000
    const int E = in_sizes[1] / 2;     // 640000

    char* ws = (char*)d_ws;
    float*     agg = (float*)ws;
    size_t off = (size_t)N * H * sizeof(float);
    _Float16* w1s = (_Float16*)(ws + off); off += 256 * H * 2;
    _Float16* w2s = (_Float16*)(ws + off); off += 128 * H * 2;
    _Float16* u1s = (_Float16*)(ws + off); off += 256 * H * 2;
    _Float16* u2s = (_Float16*)(ws + off); off += 128 * H * 2;

    (void)n_in; (void)out_size; (void)ws_size;

    const int edgeLds = 65536 + 32768 + 65536;          // 163840
    const int nodeLds = 65536 + 32768 + 65536 + 1024;   // 164864
    hipFuncSetAttribute((const void*)edge_kernel,
                        hipFuncAttributeMaxDynamicSharedMemorySize, edgeLds);
    hipFuncSetAttribute((const void*)node_kernel,
                        hipFuncAttributeMaxDynamicSharedMemorySize, nodeLds);

    {
        int n = N * H;
        zero_f32<<<(n + 255) / 256, 256, 0, stream>>>(agg, n);
    }
    swizzle_w<<<(256 * H + 255) / 256, 256, 0, stream>>>(mW1, w1s, 256);
    swizzle_w<<<(128 * H + 255) / 256, 256, 0, stream>>>(mW2, w2s, 128);
    swizzle_w<<<(256 * H + 255) / 256, 256, 0, stream>>>(uW1, u1s, 256);
    swizzle_w<<<(128 * H + 255) / 256, 256, 0, stream>>>(uW2, u2s, 128);

    // 20000 32-edge tiles, grid-stride over 8 waves per block
    edge_kernel<<<1024, 256, edgeLds, stream>>>(x, ei, ew, mW1, mb1, mb2,
                                                w1s, w2s, agg, E);

    // 625 16-node tiles
    node_kernel<<<80, 256, nodeLds, stream>>>(x, agg, ub1, ln_g, ln_b, ub2,
                                              u1s, u2s, out, N);
}
